// AVWGCN_11390253269248
// MI455X (gfx1250) — compile-verified, matrix operand order checked by reading, not run on hardware
//
#include <hip/hip_runtime.h>
#include <hip/hip_bf16.h>

typedef __attribute__((ext_vector_type(16))) _Float16 v16h;
typedef __attribute__((ext_vector_type(8)))  _Float16 v8h;
typedef __attribute__((ext_vector_type(8)))  float    v8f;

#define NN   2048   // nodes
#define DD   16     // embedding dim
#define BB   64     // batch
#define CC   64     // in channels
#define OO   64     // out channels
#define JJ   (BB*CC)   // 4096 fused (b,c) columns
#define KC   (3*CC)    // 192 fused (k,i) contraction dim

static __device__ __forceinline__ v8f wmma_f16(v16h a, v16h b, v8f c) {
  return __builtin_amdgcn_wmma_f32_16x16x32_f16(false, a, false, b, (short)0, c, false, false);
}

// ---------------------------------------------------------------------------
// K1: A = softmax(relu(E E^T)) row-wise, stored f16 row-major [N, N]
// one block (256 thr) per row
// ---------------------------------------------------------------------------
__global__ __launch_bounds__(256) void supports_softmax_kernel(
    const float* __restrict__ E, _Float16* __restrict__ A) {
  const int n = blockIdx.x;
  const int t = threadIdx.x;
  __shared__ float e[DD];
  __shared__ float vals[NN];
  __shared__ float red[8];
  if (t < DD) e[t] = E[n * DD + t];
  __syncthreads();
  float lmax = 0.0f;  // relu => max >= 0
  for (int m = t; m < NN; m += 256) {
    const float* Em = E + m * DD;
    float d = 0.f;
#pragma unroll
    for (int k = 0; k < DD; ++k) d += e[k] * Em[k];
    d = fmaxf(d, 0.f);
    vals[m] = d;
    lmax = fmaxf(lmax, d);
  }
#pragma unroll
  for (int off = 16; off; off >>= 1) lmax = fmaxf(lmax, __shfl_xor(lmax, off, 32));
  if ((t & 31) == 0) red[t >> 5] = lmax;
  __syncthreads();
  float bmax = red[0];
#pragma unroll
  for (int i = 1; i < 8; ++i) bmax = fmaxf(bmax, red[i]);
  float lsum = 0.f;
  for (int m = t; m < NN; m += 256) {
    float ex = __expf(vals[m] - bmax);
    vals[m] = ex;
    lsum += ex;
  }
#pragma unroll
  for (int off = 16; off; off >>= 1) lsum += __shfl_xor(lsum, off, 32);
  __syncthreads();
  if ((t & 31) == 0) red[t >> 5] = lsum;
  __syncthreads();
  float bsum = 0.f;
#pragma unroll
  for (int i = 0; i < 8; ++i) bsum += red[i];
  const float inv = 1.0f / bsum;
  for (int m = t; m < NN; m += 256) A[n * NN + m] = (_Float16)(vals[m] * inv);
}

// ---------------------------------------------------------------------------
// K2: feat f32 [B,N,C] -> feat_h f16 [B,N,C] and featT f16 [B,C,N] (= Xt[j,m])
// one block per (b, 64-row n-tile)
// ---------------------------------------------------------------------------
__global__ __launch_bounds__(256) void feat_convert_kernel(
    const float* __restrict__ feat, _Float16* __restrict__ feat_h,
    _Float16* __restrict__ featT) {
  const int blk = blockIdx.x;
  const int b  = blk >> 5;          // N/64 = 32 tiles
  const int n0 = (blk & 31) * 64;
  __shared__ _Float16 tile[64][CC + 8];
  for (int idx = threadIdx.x; idx < 64 * CC; idx += 256) {
    const int r = idx >> 6, c = idx & 63;
    const float v = feat[((b * NN) + (n0 + r)) * CC + c];
    const _Float16 h = (_Float16)v;
    feat_h[((b * NN) + (n0 + r)) * CC + c] = h;
    tile[r][c] = h;
  }
  __syncthreads();
  for (int idx = threadIdx.x; idx < 64 * CC; idx += 256) {
    const int c = idx >> 6, r = idx & 63;   // write contiguous along n
    featT[(b * CC + c) * NN + (n0 + r)] = tile[r][c];
  }
}

// ---------------------------------------------------------------------------
// K3: Wt[n][o][k*C+i] = sum_d E[n,d] * Wp[d,k,i,o]  (f16, o-major)
//     biasN[n][o]     = sum_d E[n,d] * bp[d,o]
// one block (256 thr) per node; weights_pool slice staged through LDS
// ---------------------------------------------------------------------------
__global__ __launch_bounds__(256) void node_weights_kernel(
    const float* __restrict__ E, const float* __restrict__ Wp,
    const float* __restrict__ bp, _Float16* __restrict__ Wt,
    float* __restrict__ biasN) {
  const int n = blockIdx.x;
  const int t = threadIdx.x;
  __shared__ float e[DD];
  __shared__ float slice[KC * OO];   // 12288 floats = 48KB
  if (t < DD) e[t] = E[n * DD + t];
  float acc[48];
#pragma unroll
  for (int s = 0; s < 48; ++s) acc[s] = 0.f;
  for (int d = 0; d < DD; ++d) {
    __syncthreads();
    for (int j = t; j < KC * OO; j += 256) slice[j] = Wp[d * (KC * OO) + j]; // [kc][o]
    __syncthreads();
    const float ed = e[d];
#pragma unroll
    for (int s = 0; s < 48; ++s) {
      const int j  = t + s * 256;   // flat (o, kc) index: j = o*KC + kc
      const int o  = j / KC;
      const int kc = j - o * KC;
      acc[s] += ed * slice[kc * OO + o];
    }
  }
#pragma unroll
  for (int s = 0; s < 48; ++s)
    Wt[n * (KC * OO) + t + s * 256] = (_Float16)acc[s];
  if (t < OO) {
    float bsum = 0.f;
#pragma unroll
    for (int d = 0; d < DD; ++d) bsum += e[d] * bp[d * OO + t];
    biasN[n * OO + t] = bsum;
  }
}

// ---------------------------------------------------------------------------
// K4: diffusion GEMM  Y[n, j] = sum_m A[n,m] * Xt[j,m]
// block tile 64x128 (4 waves, each 32 rows x 64 cols), K stepped by 32
// stores Y as f16 [B,N,C]; optionally also transposed Yt[j,n] for next hop
// ---------------------------------------------------------------------------
__global__ __launch_bounds__(128) void diff_gemm_kernel(
    const _Float16* __restrict__ A, const _Float16* __restrict__ Xt,
    _Float16* __restrict__ Ybnc, _Float16* __restrict__ Yt, int writeYt) {
  const int lane = threadIdx.x & 31;
  const int w    = threadIdx.x >> 5;
  const int lrow = lane & 15;
  const int lsel = lane >> 4;
  const int n_base = blockIdx.x * 64 + (w >> 1) * 32;
  const int j_base = blockIdx.y * 128 + (w & 1) * 64;

  v8f c[2][4];
#pragma unroll
  for (int i = 0; i < 2; ++i)
#pragma unroll
    for (int jt = 0; jt < 4; ++jt) c[i][jt] = (v8f)0.f;

  const _Float16* Ar0 = A + (n_base + lrow) * NN + lsel * 16;
  const _Float16* Ar1 = Ar0 + 16 * NN;
  const _Float16* Br0 = Xt + (j_base +  0 + lrow) * NN + lsel * 16;
  const _Float16* Br1 = Br0 + 16 * NN;
  const _Float16* Br2 = Br1 + 16 * NN;
  const _Float16* Br3 = Br2 + 16 * NN;

  for (int k0 = 0; k0 < NN; k0 += 32) {
    __builtin_prefetch(Ar0 + k0 + 256, 0, 0);
    __builtin_prefetch(Ar1 + k0 + 256, 0, 0);
    const v16h a0 = *(const v16h*)(Ar0 + k0);
    const v16h a1 = *(const v16h*)(Ar1 + k0);
    const v16h b0 = *(const v16h*)(Br0 + k0);
    const v16h b1 = *(const v16h*)(Br1 + k0);
    const v16h b2 = *(const v16h*)(Br2 + k0);
    const v16h b3 = *(const v16h*)(Br3 + k0);
    c[0][0] = wmma_f16(a0, b0, c[0][0]);
    c[0][1] = wmma_f16(a0, b1, c[0][1]);
    c[0][2] = wmma_f16(a0, b2, c[0][2]);
    c[0][3] = wmma_f16(a0, b3, c[0][3]);
    c[1][0] = wmma_f16(a1, b0, c[1][0]);
    c[1][1] = wmma_f16(a1, b1, c[1][1]);
    c[1][2] = wmma_f16(a1, b2, c[1][2]);
    c[1][3] = wmma_f16(a1, b3, c[1][3]);
  }

#pragma unroll
  for (int i = 0; i < 2; ++i) {
    const int nst = n_base + i * 16 + lsel * 8;   // first of 8 consecutive rows
#pragma unroll
    for (int jt = 0; jt < 4; ++jt) {
      const int j  = j_base + jt * 16 + lrow;
      const int bb = j >> 6;      // batch
      const int cc = j & 63;      // channel
      v8h hv;
#pragma unroll
      for (int r = 0; r < 8; ++r) {
        const _Float16 h = (_Float16)c[i][jt][r];
        hv[r] = h;
        Ybnc[((bb * NN) + nst + r) * CC + cc] = h;
      }
      if (writeYt) *(v8h*)(Yt + j * NN + nst) = hv;
    }
  }
}

// ---------------------------------------------------------------------------
// K5: per-node GEMM  out[b,n,o] = sum_{kc} xg[b, kc] * Wt[n][o][kc] + bias[n,o]
// xg k=0: feat, k=1: y1, k=2: 2*y2 - feat (built on the fly, packed f16)
// one block (4 waves) per node; wave handles 16 b-rows x 64 o-cols
// ---------------------------------------------------------------------------
__global__ __launch_bounds__(128) void final_gemm_kernel(
    const _Float16* __restrict__ feat_h, const _Float16* __restrict__ y1_h,
    const _Float16* __restrict__ y2_h, const _Float16* __restrict__ Wt,
    const float* __restrict__ biasN, float* __restrict__ out) {
  const int n    = blockIdx.x;
  const int lane = threadIdx.x & 31;
  const int w    = threadIdx.x >> 5;
  const int lrow = lane & 15;
  const int lsel = lane >> 4;
  const int b0   = w * 16;

  v8f c[4];
#pragma unroll
  for (int ot = 0; ot < 4; ++ot) c[ot] = (v8f)0.f;

  const int rowbase = ((b0 + lrow) * NN + n) * CC;  // [B,N,C] row for this lane

#pragma unroll
  for (int kk = 0; kk < 6; ++kk) {           // 6 chunks of 32 over KC=192
    const int k  = kk >> 1;
    const int io = (kk & 1) * 32 + lsel * 16;
    v16h a;
    if (k == 0) {
      a = *(const v16h*)(feat_h + rowbase + io);
    } else if (k == 1) {
      a = *(const v16h*)(y1_h + rowbase + io);
    } else {
      const v16h y2 = *(const v16h*)(y2_h + rowbase + io);
      const v16h f0 = *(const v16h*)(feat_h + rowbase + io);
      const _Float16 two = (_Float16)2.0f;
#pragma unroll
      for (int t = 0; t < 16; ++t) a[t] = two * y2[t] - f0[t];
    }
    const _Float16* wb = Wt + n * (KC * OO) + kk * 32 + lsel * 16;
#pragma unroll
    for (int ot = 0; ot < 4; ++ot) {
      const v16h bf = *(const v16h*)(wb + (ot * 16 + lrow) * KC);
      c[ot] = wmma_f16(a, bf, c[ot]);
    }
  }

#pragma unroll
  for (int ot = 0; ot < 4; ++ot) {
    const int o = ot * 16 + lrow;
    const float bv = biasN[n * OO + o];
#pragma unroll
    for (int r = 0; r < 8; ++r) {
      const int bb = b0 + lsel * 8 + r;
      out[((bb * NN) + n) * OO + o] = c[ot][r] + bv;
    }
  }
}

// ---------------------------------------------------------------------------
extern "C" void kernel_launch(void* const* d_in, const int* in_sizes, int n_in,
                              void* d_out, int out_size, void* d_ws, size_t ws_size,
                              hipStream_t stream) {
  const float* E    = (const float*)d_in[0];   // [N, D]
  const float* feat = (const float*)d_in[1];   // [B, N, C]
  const float* Wp   = (const float*)d_in[2];   // [D, K, C, O]
  const float* bp   = (const float*)d_in[3];   // [D, O]
  float* out = (float*)d_out;                  // [B, N, O]

  // workspace partition (each region 256B-aligned)
  char* p = (char*)d_ws;
  auto take = [&](size_t bytes) {
    char* q = p;
    p += (bytes + 255) & ~(size_t)255;
    return q;
  };
  _Float16* A_h    = (_Float16*)take((size_t)NN * NN * 2);        // 8 MB
  _Float16* featT  = (_Float16*)take((size_t)JJ * NN * 2);        // 16 MB
  _Float16* feat_h = (_Float16*)take((size_t)BB * NN * CC * 2);   // 16 MB
  _Float16* y1t    = (_Float16*)take((size_t)JJ * NN * 2);        // 16 MB
  _Float16* y1_h   = (_Float16*)take((size_t)BB * NN * CC * 2);   // 16 MB
  _Float16* y2_h   = (_Float16*)take((size_t)BB * NN * CC * 2);   // 16 MB
  _Float16* Wt     = (_Float16*)take((size_t)NN * KC * OO * 2);   // 48 MB
  float*    biasN  = (float*)   take((size_t)NN * OO * 4);        // 0.5 MB

  // K1: adaptive adjacency (softmax of relu(E E^T)) -> f16
  supports_softmax_kernel<<<NN, 256, 0, stream>>>(E, A_h);
  // K2: feature convert + transpose
  feat_convert_kernel<<<BB * (NN / 64), 256, 0, stream>>>(feat, feat_h, featT);
  // K3: node-conditioned weights / bias
  node_weights_kernel<<<NN, 256, 0, stream>>>(E, Wp, bp, Wt, biasN);
  // K4a: y1 = A @ feat   (also write y1 transposed for the next hop)
  diff_gemm_kernel<<<dim3(NN / 64, JJ / 128), 128, 0, stream>>>(
      A_h, featT, y1_h, y1t, 1);
  // K4b: y2 = A @ y1
  diff_gemm_kernel<<<dim3(NN / 64, JJ / 128), 128, 0, stream>>>(
      A_h, y1t, y2_h, nullptr, 0);
  // K5: per-node contraction with (feat, y1, 2*y2 - feat) + bias
  final_gemm_kernel<<<NN, 128, 0, stream>>>(feat_h, y1_h, y2_h, Wt, biasN, out);
}
